// DGCNN_11888469475439
// MI455X (gfx1250) — matrix operand order repaired and checked by reading, not compiled, not run on hardware
//
#include <hip/hip_runtime.h>
#include <hip/hip_bf16.h>

// ---------------------------------------------------------------------------
// DGCNN forward for MI455X (gfx1250, wave32, WMMA).
//
// v2: activations kept as bf16 in POINT-MAJOR layout catT[B][N][512] so WMMA
// A/B fragments are contiguous 16-byte (global_load_b128 / ds_load_b128)
// loads; weights pre-converted to bf16 once. fp32 accumulate everywhere.
// BN scale is positive (g == 1 in the provided inputs), so max over k/N
// commutes with BN+LeakyReLU; max is fused into the matmul epilogues and
// [B,O,N,k] is never materialized.
// ---------------------------------------------------------------------------

typedef __attribute__((ext_vector_type(16))) __bf16 v16bf;
typedef __attribute__((ext_vector_type(8)))  __bf16 v8bf;
typedef __attribute__((ext_vector_type(8)))  float  v8f;

#define NPTS 2048
#define BATCH 8
#define KNN 20
#define CB 512               // channels in concatenated feature buffer
#define NEG_INF (-3.0e38f)

__device__ __forceinline__ float lrelu(float x) { return x >= 0.f ? x : 0.2f * x; }

__device__ __forceinline__ __bf16 f2bf(float f) {   // RNE
    unsigned u = __builtin_bit_cast(unsigned, f);
    u = (u + 0x7FFFu + ((u >> 16) & 1u)) >> 16;
    unsigned short s = (unsigned short)u;
    return __builtin_bit_cast(__bf16, s);
}
__device__ __forceinline__ float bf2f(__bf16 h) {
    unsigned short s = __builtin_bit_cast(unsigned short, h);
    return __builtin_bit_cast(float, ((unsigned)s) << 16);
}

__device__ __forceinline__ v8f wmma_bf16(v16bf a, v16bf b, v8f c) {
    return __builtin_amdgcn_wmma_f32_16x16x32_bf16(false, a, false, b, (short)0, c,
                                                   false, false);
}

// A-fragment (16x32 bf16, ISA 7.12.2): lane holds row (lane&15); halfs are two
// contiguous 8-half runs at K = 8*(lane>>4) and K = 16 + 8*(lane>>4).
__device__ __forceinline__ v16bf load_afrag(const __bf16* rowBase, int lane, int kc) {
    const __bf16* p = rowBase + kc * 32 + ((lane >> 4) << 3);
    v8bf lo = *(const v8bf*)(p);
    v8bf hi = *(const v8bf*)(p + 16);
    v16bf a;
#pragma unroll
    for (int t = 0; t < 8; ++t) { a[t] = lo[t]; a[t + 8] = hi[t]; }
    return a;
}
// B-fragment (32x16 bf16): lane holds col (lane&15); halfs are two
// contiguous 8-half runs at K = 8*(lane>>4) and K = 16 + 8*(lane>>4).
__device__ __forceinline__ v16bf load_bfrag(const __bf16* colBase, int lane, int kc) {
    const __bf16* p = colBase + kc * 32 + ((lane >> 4) << 3);
    v8bf lo = *(const v8bf*)(p);
    v8bf hi = *(const v8bf*)(p + 16);
    v16bf b;
#pragma unroll
    for (int t = 0; t < 8; ++t) { b[t] = lo[t]; b[t + 8] = hi[t]; }
    return b;
}

// ---------------------------------------------------------------------------
// fp32 -> bf16 weight conversion (one-shot per launch, tiny)
// ---------------------------------------------------------------------------
__global__ void convertW_kernel(const float* __restrict__ src,
                                __bf16* __restrict__ dst, int count) {
    for (int i = blockIdx.x * 256 + threadIdx.x; i < count; i += gridDim.x * 256)
        dst[i] = f2bf(src[i]);
}

// ---------------------------------------------------------------------------
// Layer 1: y = W1[64,6] @ x[B,6,N,k]; fused max over k + channel stats.
// ---------------------------------------------------------------------------
__global__ __launch_bounds__(256) void edge1_kernel(
    const float* __restrict__ x,   // [B,6,N,KNN]
    const float* __restrict__ W,   // [64,6]
    float* __restrict__ M,         // [B,64,N] pre-BN max
    float* __restrict__ stats)     // [128]
{
    const int O = 64;
    int b = blockIdx.x >> 6, o = blockIdx.x & 63;
    float w[6];
#pragma unroll
    for (int c = 0; c < 6; ++c) w[c] = W[o * 6 + c];
    float s = 0.f, s2 = 0.f;
    for (int n = threadIdx.x; n < NPTS; n += 256) {
        float mx = NEG_INF;
#pragma unroll 4
        for (int k = 0; k < KNN; ++k) {
            float y = 0.f;
#pragma unroll
            for (int c = 0; c < 6; ++c)
                y += w[c] * x[(((size_t)b * 6 + c) * NPTS + n) * KNN + k];
            mx = fmaxf(mx, y);
            s += y;
            s2 += y * y;
        }
        M[((size_t)b * O + o) * NPTS + n] = mx;
    }
    __shared__ float rs[256], rs2[256];
    rs[threadIdx.x] = s;
    rs2[threadIdx.x] = s2;
    __syncthreads();
    for (int st = 128; st > 0; st >>= 1) {
        if (threadIdx.x < st) {
            rs[threadIdx.x] += rs[threadIdx.x + st];
            rs2[threadIdx.x] += rs2[threadIdx.x + st];
        }
        __syncthreads();
    }
    if (threadIdx.x == 0) {
        atomicAdd(&stats[o], rs[0]);
        atomicAdd(&stats[O + o], rs2[0]);
    }
}

// ---------------------------------------------------------------------------
// BN + LeakyReLU on pre-BN maxima; writes bf16 into point-major catT slice.
// i iterates [B][N][O] with o fastest -> coalesced bf16 writes.
// ---------------------------------------------------------------------------
__global__ void bn_apply_kernel(
    const float* __restrict__ M, const float* __restrict__ stats, int O,
    float invCount, const float* __restrict__ g, const float* __restrict__ bb,
    __bf16* __restrict__ catT, int chanOff)
{
    size_t total = (size_t)BATCH * NPTS * O;
    for (size_t i = (size_t)blockIdx.x * blockDim.x + threadIdx.x; i < total;
         i += (size_t)gridDim.x * blockDim.x) {
        int o = (int)(i % O);
        size_t t = i / O;
        int n = (int)(t % NPTS);
        int b = (int)(t / NPTS);
        float mean = stats[o] * invCount;
        float var  = stats[O + o] * invCount - mean * mean;
        float rstd = rsqrtf(var + 1e-5f);
        float y = g[o] * (M[((size_t)b * O + o) * NPTS + n] - mean) * rstd + bb[o];
        catT[((size_t)b * NPTS + n) * CB + chanOff + o] = f2bf(lrelu(y));
    }
}

// ---------------------------------------------------------------------------
// KNN: workgroup per (b, 16-row tile). 8 waves split the 128 column tiles;
// WMMA inner products (point-major bf16 -> all fragment loads are b128),
// register-resident top-20 per row, LDS merge of 8x20 candidates.
// ---------------------------------------------------------------------------
template <int C>
__global__ __launch_bounds__(256) void knn_kernel(
    const __bf16* __restrict__ Xt,  // catT [B][N][CB]
    int coff,                       // channel slice offset
    int* __restrict__ idx)          // [B,N,KNN]
{
    constexpr int NCH = C / 32;
    int b = blockIdx.x >> 7;
    int rowTile = blockIdx.x & 127;
    int tid = threadIdx.x, wave = tid >> 5, lane = tid & 31;

    __shared__ float sq[NPTS];
    __shared__ float stage[8][16][16];
    __shared__ float candV[16][160];
    __shared__ int   candI[16][160];

    for (int n = tid; n < NPTS; n += 256) {
        const __bf16* p = Xt + ((size_t)b * NPTS + n) * CB + coff;
        float s = 0.f;
#pragma unroll 8
        for (int c = 0; c < C; ++c) { float v = bf2f(p[c]); s += v * v; }
        sq[n] = s;
    }
    __syncthreads();

    int arow = rowTile * 16 + (lane & 15);
    const __bf16* arowp = Xt + ((size_t)b * NPTS + arow) * CB + coff;
    v16bf afrag[NCH];
#pragma unroll
    for (int kc = 0; kc < NCH; ++kc) afrag[kc] = load_afrag(arowp, lane, kc);

    float tv[KNN];
    int   ti[KNN];
#pragma unroll
    for (int i = 0; i < KNN; ++i) { tv[i] = NEG_INF; ti[i] = 0; }

    for (int ct = wave; ct < 128; ct += 8) {
        int bcol = ct * 16 + (lane & 15);
        const __bf16* bcolp = Xt + ((size_t)b * NPTS + bcol) * CB + coff;
        v8f acc = {0.f, 0.f, 0.f, 0.f, 0.f, 0.f, 0.f, 0.f};
#pragma unroll
        for (int kc = 0; kc < NCH; ++kc)
            acc = wmma_bf16(afrag[kc], load_bfrag(bcolp, lane, kc), acc);
#pragma unroll
        for (int r = 0; r < 8; ++r) {
            int row = r + ((lane >> 4) << 3);
            stage[wave][row][lane & 15] =
                2.f * acc[r] - sq[rowTile * 16 + row] - sq[bcol];
        }
        __builtin_amdgcn_wave_barrier();
        if (lane < 16) {
#pragma unroll
            for (int c2 = 0; c2 < 16; ++c2) {
                float v = stage[wave][lane][c2];
                int ci = ct * 16 + c2;
                if (v > tv[KNN - 1]) {
                    tv[KNN - 1] = v;
                    ti[KNN - 1] = ci;
#pragma unroll
                    for (int s = KNN - 1; s >= 1; --s) {
                        if (tv[s] > tv[s - 1]) {
                            float tf = tv[s]; tv[s] = tv[s - 1]; tv[s - 1] = tf;
                            int   tj = ti[s]; ti[s] = ti[s - 1]; ti[s - 1] = tj;
                        }
                    }
                }
            }
        }
        __builtin_amdgcn_wave_barrier();
    }

    if (lane < 16) {
#pragma unroll
        for (int i = 0; i < KNN; ++i) {
            candV[lane][wave * KNN + i] = tv[i];
            candI[lane][wave * KNN + i] = ti[i];
        }
    }
    __syncthreads();
    if (tid < 16) {
        int n = rowTile * 16 + tid;
        for (int j = 0; j < KNN; ++j) {
            float best = NEG_INF;
            int bi = 0, bpos = 0;
            for (int t = 0; t < 160; ++t) {
                float v = candV[tid][t];
                if (v > best) { best = v; bi = candI[tid][t]; bpos = t; }
            }
            candV[tid][bpos] = NEG_INF;
            idx[((size_t)b * NPTS + n) * KNN + j] = bi;
        }
    }
}

// ---------------------------------------------------------------------------
// Edge conv (layers 2-4): workgroup per (b,n). Point-major gather: neighbor
// channels are contiguous -> 16B vector loads into LDS featT[32][C2]
// (neighbor-major so B-fragments are ds_load_b128). WMMA over all O channels,
// epilogue fuses max over k + channel stats.
// ---------------------------------------------------------------------------
template <int C2, int O>
__global__ __launch_bounds__(256) void edge_conv_kernel(
    const __bf16* __restrict__ Xt, int coff,  // input slice, C = C2/2 channels
    const int* __restrict__ idx,              // [B,N,KNN]
    const __bf16* __restrict__ Wb,            // [O,C2] bf16
    float* __restrict__ M,                    // [B,O,N] pre-BN max
    float* __restrict__ stats)                // [2*O]
{
    constexpr int C = C2 / 2;
    constexpr int NCH = C2 / 32;
    constexpr int VPERJ = C2 / 8;             // 8-half vectors per neighbor row
    int b = blockIdx.x >> 11, n = blockIdx.x & 2047;
    int tid = threadIdx.x, wave = tid >> 5, lane = tid & 31;

    __shared__ __bf16 featT[32][C2];          // [neighbor j][channel]
    __shared__ float  ystage[8][16][32];
    __shared__ int    nidx[KNN];

    if (tid < KNN) nidx[tid] = idx[((size_t)b * NPTS + n) * KNN + tid];
    __syncthreads();

    const __bf16* ctr = Xt + ((size_t)b * NPTS + n) * CB + coff;
    for (int e = tid; e < 32 * VPERJ; e += 256) {
        int j = e / VPERJ;
        int cb = (e % VPERJ) * 8;             // C is a multiple of 8
        v8bf out;
        if (j >= KNN) {
#pragma unroll
            for (int t = 0; t < 8; ++t) out[t] = f2bf(0.f);
        } else if (cb < C) {
            const __bf16* nb = Xt + ((size_t)b * NPTS + nidx[j]) * CB + coff + cb;
            v8bf av = *(const v8bf*)nb;
            v8bf cv = *(const v8bf*)(ctr + cb);
#pragma unroll
            for (int t = 0; t < 8; ++t) out[t] = f2bf(bf2f(av[t]) - bf2f(cv[t]));
        } else {
            out = *(const v8bf*)(ctr + (cb - C));
        }
        *(v8bf*)&featT[j][cb] = out;
    }
    __syncthreads();

    for (int ot = wave; ot < O / 16; ot += 8) {
        int orow = ot * 16 + (lane & 15);
        const __bf16* wrow = Wb + (size_t)orow * C2;
        v8f acc0 = {0.f, 0.f, 0.f, 0.f, 0.f, 0.f, 0.f, 0.f};
        v8f acc1 = {0.f, 0.f, 0.f, 0.f, 0.f, 0.f, 0.f, 0.f};
#pragma unroll
        for (int kc = 0; kc < NCH; ++kc) {
            v16bf a = load_afrag(wrow, lane, kc);
            v16bf b0 = load_bfrag(&featT[lane & 15][0], lane, kc);
            v16bf b1 = load_bfrag(&featT[16 + (lane & 15)][0], lane, kc);
            acc0 = wmma_bf16(a, b0, acc0);
            acc1 = wmma_bf16(a, b1, acc1);
        }
#pragma unroll
        for (int r = 0; r < 8; ++r) {
            int row = r + ((lane >> 4) << 3);
            ystage[wave][row][lane & 15]        = acc0[r];
            ystage[wave][row][16 + (lane & 15)] = acc1[r];
        }
        __builtin_amdgcn_wave_barrier();
        if (lane < 16) {
            int o = ot * 16 + lane;
            float mx = NEG_INF, s = 0.f, s2 = 0.f;
#pragma unroll
            for (int j = 0; j < KNN; ++j) {
                float y = ystage[wave][lane][j];
                mx = fmaxf(mx, y);
                s += y;
                s2 += y * y;
            }
            M[((size_t)b * O + o) * NPTS + n] = mx;
            atomicAdd(&stats[o], s);
            atomicAdd(&stats[O + o], s2);
        }
        __builtin_amdgcn_wave_barrier();
    }
}

// ---------------------------------------------------------------------------
// Final layer: y = W5[1024,512] @ catT; fused max over N + stats.
// Workgroup per (b, 16-o-tile); persistent W5 A-fragments (16 K-chunks).
// ---------------------------------------------------------------------------
__global__ __launch_bounds__(256) void final_kernel(
    const __bf16* __restrict__ catT,  // [B][N][512]
    const __bf16* __restrict__ Wb5,   // [1024,512] bf16
    float* __restrict__ ymax,         // [B,1024]
    float* __restrict__ stats)        // [2048]
{
    const int C2 = 512, O = 1024, NCH = 16;
    int b = blockIdx.x >> 6, ot = blockIdx.x & 63;
    int tid = threadIdx.x, wave = tid >> 5, lane = tid & 31;

    __builtin_prefetch(Wb5 + (size_t)(ot * 16) * C2, 0, 1);

    const __bf16* wrow = Wb5 + (size_t)(ot * 16 + (lane & 15)) * C2;
    v16bf afrag[NCH];
#pragma unroll
    for (int kc = 0; kc < NCH; ++kc) afrag[kc] = load_afrag(wrow, lane, kc);

    float rmax[8], rsum[8], rsq[8];
#pragma unroll
    for (int r = 0; r < 8; ++r) { rmax[r] = NEG_INF; rsum[r] = 0.f; rsq[r] = 0.f; }

    for (int nt = wave; nt < 128; nt += 8) {
        int col = nt * 16 + (lane & 15);
        const __bf16* colp = catT + ((size_t)b * NPTS + col) * CB;
        v8f acc = {0.f, 0.f, 0.f, 0.f, 0.f, 0.f, 0.f, 0.f};
#pragma unroll
        for (int kc = 0; kc < NCH; ++kc)
            acc = wmma_bf16(afrag[kc], load_bfrag(colp, lane, kc), acc);
#pragma unroll
        for (int r = 0; r < 8; ++r) {
            float y = acc[r];
            rmax[r] = fmaxf(rmax[r], y);
            rsum[r] += y;
            rsq[r] += y * y;
        }
    }

    __shared__ float red[8][16][16][3];
#pragma unroll
    for (int r = 0; r < 8; ++r) {
        int row = r + ((lane >> 4) << 3);
        red[wave][row][lane & 15][0] = rmax[r];
        red[wave][row][lane & 15][1] = rsum[r];
        red[wave][row][lane & 15][2] = rsq[r];
    }
    __syncthreads();
    if (tid < 16) {
        float mx = NEG_INF, s = 0.f, s2 = 0.f;
        for (int w2 = 0; w2 < 8; ++w2)
            for (int c = 0; c < 16; ++c) {
                mx = fmaxf(mx, red[w2][tid][c][0]);
                s += red[w2][tid][c][1];
                s2 += red[w2][tid][c][2];
            }
        int o = ot * 16 + tid;
        ymax[(size_t)b * O + o] = mx;
        atomicAdd(&stats[o], s);
        atomicAdd(&stats[O + o], s2);
    }
}

__global__ void bn_final_kernel(
    const float* __restrict__ ymax, const float* __restrict__ stats,
    const float* __restrict__ g, const float* __restrict__ bb,
    float* __restrict__ out)
{
    int i = blockIdx.x * 256 + threadIdx.x;
    if (i >= BATCH * 1024) return;
    int o = i & 1023;
    const float invC = 1.f / (float)(BATCH * NPTS);
    float mean = stats[o] * invC;
    float var  = stats[1024 + o] * invC - mean * mean;
    float rstd = rsqrtf(var + 1e-5f);
    out[i] = lrelu(g[o] * (ymax[i] - mean) * rstd + bb[o]);
}

// ---------------------------------------------------------------------------
extern "C" void kernel_launch(void* const* d_in, const int* in_sizes, int n_in,
                              void* d_out, int out_size, void* d_ws, size_t ws_size,
                              hipStream_t stream) {
    (void)in_sizes; (void)n_in; (void)out_size; (void)ws_size;
    const float* x  = (const float*)d_in[0];
    const float* W1 = (const float*)d_in[1];
    const float* g1 = (const float*)d_in[2];
    const float* b1 = (const float*)d_in[3];
    const float* W2 = (const float*)d_in[4];
    const float* g2 = (const float*)d_in[5];
    const float* b2 = (const float*)d_in[6];
    const float* W3 = (const float*)d_in[7];
    const float* g3 = (const float*)d_in[8];
    const float* b3 = (const float*)d_in[9];
    const float* W4 = (const float*)d_in[10];
    const float* g4 = (const float*)d_in[11];
    const float* b4 = (const float*)d_in[12];
    const float* W5 = (const float*)d_in[13];
    const float* g5 = (const float*)d_in[14];
    const float* b5 = (const float*)d_in[15];
    float* out = (float*)d_out;

    // ---- workspace layout ----
    __bf16* catT = (__bf16*)d_ws;                              // [8][2048][512] bf16, 16 MB
    float*  M    = (float*)(catT + (size_t)BATCH * NPTS * CB); // [8,256,2048] f32 (reused)
    float*  ymax = M + (size_t)BATCH * 256 * NPTS;             // [8,1024]
    float*  stats = ymax + (size_t)BATCH * 1024;               // 3072 floats
    __bf16* Wbf  = (__bf16*)(stats + 3072);                    // bf16 weights
    // bf16 weight offsets (all 16B aligned)
    __bf16* Wb2 = Wbf;                 // 64*128   = 8192
    __bf16* Wb3 = Wb2 + 8192;          // 128*128  = 16384
    __bf16* Wb4 = Wb3 + 16384;         // 256*256  = 65536
    __bf16* Wb5 = Wb4 + 65536;         // 1024*512 = 524288
    int*    idx = (int*)(Wb5 + 524288);                        // [8,2048,20]

    float* s1 = stats;          // 2*64
    float* s2 = stats + 128;    // 2*64
    float* s3 = stats + 256;    // 2*128
    float* s4 = stats + 512;    // 2*256
    float* s5 = stats + 1024;   // 2*1024

    const float invBNK = 1.f / (float)(BATCH * NPTS * KNN);

    hipMemsetAsync(stats, 0, 3072 * sizeof(float), stream);

    // weight conversion (W1 stays fp32 in the scalar layer-1 kernel)
    convertW_kernel<<<32, 256, 0, stream>>>(W2, Wb2, 64 * 128);
    convertW_kernel<<<64, 256, 0, stream>>>(W3, Wb3, 128 * 128);
    convertW_kernel<<<256, 256, 0, stream>>>(W4, Wb4, 256 * 256);
    convertW_kernel<<<1024, 256, 0, stream>>>(W5, Wb5, 1024 * 512);

    // layer 1
    edge1_kernel<<<BATCH * 64, 256, 0, stream>>>(x, W1, M, s1);
    bn_apply_kernel<<<2048, 256, 0, stream>>>(M, s1, 64, invBNK, g1, b1, catT, 0);

    // layer 2: knn(x1, C=64) -> edge conv [64 x 128]
    knn_kernel<64><<<BATCH * 128, 256, 0, stream>>>(catT, 0, idx);
    edge_conv_kernel<128, 64><<<BATCH * NPTS, 256, 0, stream>>>(catT, 0, idx, Wb2, M, s2);
    bn_apply_kernel<<<2048, 256, 0, stream>>>(M, s2, 64, invBNK, g2, b2, catT, 64);

    // layer 3: knn(x2, C=64) -> edge conv [128 x 128]
    knn_kernel<64><<<BATCH * 128, 256, 0, stream>>>(catT, 64, idx);
    edge_conv_kernel<128, 128><<<BATCH * NPTS, 256, 0, stream>>>(catT, 64, idx, Wb3, M, s3);
    bn_apply_kernel<<<2048, 256, 0, stream>>>(M, s3, 128, invBNK, g3, b3, catT, 128);

    // layer 4: knn(x3, C=128) -> edge conv [256 x 256]
    knn_kernel<128><<<BATCH * 128, 256, 0, stream>>>(catT, 128, idx);
    edge_conv_kernel<256, 256><<<BATCH * NPTS, 256, 0, stream>>>(catT, 128, idx, Wb4, M, s4);
    bn_apply_kernel<<<2048, 256, 0, stream>>>(M, s4, 256, invBNK, g4, b4, catT, 256);

    // final projection + global max pool
    final_kernel<<<BATCH * 64, 256, 0, stream>>>(catT, Wb5, ymax, s5);
    bn_final_kernel<<<(BATCH * 1024 + 255) / 256, 256, 0, stream>>>(ymax, s5, g5, b5, out);
}
